// DeformableTransformerACTR_53343493816683
// MI455X (gfx1250) — matrix operand order repaired and checked by reading, not compiled, not run on hardware
//
#include <hip/hip_runtime.h>
#include <hip/hip_bf16.h>
#include <stdint.h>

// ---------------- problem constants ----------------
#define Bb   2
#define NQ   10000
#define Cc   256
#define NH   8
#define DH   32
#define Ll   4
#define Pp   4
#define DFF  1024
#define Ss   19560          // 92*160 + 46*80 + 23*40 + 12*20
#define BS   (Bb*Ss)        // 39120
#define BQ   (Bb*NQ)        // 20000

typedef __attribute__((ext_vector_type(16))) __bf16       v16bf;
typedef __attribute__((ext_vector_type(8)))  float        v8f;
typedef __attribute__((ext_vector_type(4)))  unsigned int v4u;

union Frag16 { v16bf v; v4u q[2]; };

// ---------------------------------------------------------------------------
// Pack fp32 weight [K,N] row-major into per-lane WMMA B-fragment layout.
// Flat packed elem p = ((nt*Ktiles + kt)*32 + lane)*16 + e ; k = kt*32 + (lane>=16)*8 + (e&7) + (e>>3)*16
// ---------------------------------------------------------------------------
__global__ void pack_b_kernel(const float* __restrict__ W, __hip_bfloat16* __restrict__ Bp,
                              int K, int N) {
    int t = blockIdx.x * 256 + threadIdx.x;
    if (t >= K * N) return;
    int e    = t & 15;
    int lane = (t >> 4) & 31;
    int tile = t >> 9;
    int Ktiles = K >> 5;
    int kt = tile % Ktiles;
    int nt = tile / Ktiles;
    int k  = kt * 32 + ((lane >> 4) & 1) * 8 + (e & 7) + (e >> 3) * 16;
    int n  = nt * 16 + (lane & 15);
    Bp[t] = __float2bfloat16(W[(size_t)k * N + n]);
}

// a[i] (+ b[i]) -> bf16
__global__ void addcvt_kernel(const float* __restrict__ a, const float* __restrict__ b,
                              __hip_bfloat16* __restrict__ o, int n) {
    int i = blockIdx.x * 256 + threadIdx.x;
    if (i >= n) return;
    float v = a[i];
    if (b) v += b[i];
    o[i] = __float2bfloat16(v);
}

// ---------------------------------------------------------------------------
// bf16 GEMM with f32 accumulation via v_wmma_f32_16x16x32_bf16.
// A: bf16 row-major [M,K].  Bp: packed B fragments.  Epilogue: +bias, opt ReLU,
// output fp32 or bf16 row-major [M,N].
// Block = 256 threads = 8 waves (2 M-tiles x 4 N-tiles => 32x64 block tile).
// ---------------------------------------------------------------------------
template<bool RELU, bool OUTBF>
__global__ __launch_bounds__(256)
void gemm_bf16_kernel(const __hip_bfloat16* __restrict__ A,
                      const __hip_bfloat16* __restrict__ Bp,
                      const float* __restrict__ bias,
                      void* __restrict__ Cout,
                      int M, int K, int N) {
    const int wid   = threadIdx.x >> 5;
    const int lane  = threadIdx.x & 31;
    const int tileM = blockIdx.y * 2 + (wid >> 2);
    const int tileN = blockIdx.x * 4 + (wid & 3);
    if (tileN * 16 >= N) return;

    const int Ktiles = K >> 5;
    int row  = tileM * 16 + (lane & 15);
    int arow = row < (M - 1) ? row : (M - 1);          // clamp loads; stores guarded
    const int kb = (lane >> 4) * 8;                    // A-fragment lane K base (0 or 8)
    const __hip_bfloat16* Abase = A + (size_t)arow * K + kb;
    const v4u* Bbase = (const v4u*)Bp + ((size_t)tileN * Ktiles * 32 + lane) * 2;

    v8f acc = {};
    for (int kt = 0; kt < Ktiles; ++kt) {
        Frag16 af, bf;
        af.q[0] = *(const v4u*)(Abase + kt * 32);       // k = kb .. kb+7
        af.q[1] = *(const v4u*)(Abase + kt * 32 + 16);  // k = kb+16 .. kb+23
        const v4u* pb = Bbase + (size_t)kt * 64;        // 32 lanes * 32B per k-tile
        bf.q[0] = pb[0];
        bf.q[1] = pb[1];
        acc = __builtin_amdgcn_wmma_f32_16x16x32_bf16(
            false, af.v, false, bf.v, (short)0, acc, false, false);
    }

    const int n     = tileN * 16 + (lane & 15);
    const int mbase = tileM * 16 + ((lane >> 4) << 3);
    const float bv  = bias ? bias[n] : 0.0f;
#pragma unroll
    for (int v = 0; v < 8; ++v) {
        int m = mbase + v;
        if (m < M) {
            float val = acc[v] + bv;
            if (RELU) val = fmaxf(val, 0.0f);
            if constexpr (OUTBF)
                ((__hip_bfloat16*)Cout)[(size_t)m * N + n] = __float2bfloat16(val);
            else
                ((float*)Cout)[(size_t)m * N + n] = val;
        }
    }
}

// ---------------------------------------------------------------------------
// Multi-scale deformable attention core. One block per (b,q); one wave per
// head; lane = channel (DH=32 == wave32) -> coalesced 128B gathers (L2-resident).
// Writes bf16 (consumed only by the Wo GEMM).
// ---------------------------------------------------------------------------
__global__ __launch_bounds__(256)
void msdeform_kernel(const float* __restrict__ value,     // [B,S,NH,DH]
                     const float* __restrict__ off,       // [BQ, NH*L*P*2]
                     const float* __restrict__ attnlog,   // [BQ, NH*L*P]
                     const float* __restrict__ qref,      // [B,NQ,2]
                     const float* __restrict__ vrat,      // [B,L,2]
                     __hip_bfloat16* __restrict__ outbf)  // [BQ, C]
{
    const int Hs[4] = {92, 46, 23, 12};
    const int Ws[4] = {160, 80, 40, 20};
    const int St[4] = {0, 14720, 18400, 19320};

    const int bq   = blockIdx.x;
    const int b    = bq / NQ;
    const int h    = threadIdx.x >> 5;
    const int lane = threadIdx.x & 31;

    // softmax over the 16 (level,point) logits of this head (uniform per wave)
    const float* lg = attnlog + ((size_t)bq * NH + h) * (Ll * Pp);
    float p[16], mx = -1e30f;
#pragma unroll
    for (int j = 0; j < 16; ++j) { p[j] = lg[j]; mx = fmaxf(mx, p[j]); }
    float sum = 0.0f;
#pragma unroll
    for (int j = 0; j < 16; ++j) { p[j] = __expf(p[j] - mx); sum += p[j]; }
    const float inv = 1.0f / sum;

    const float rx = qref[(size_t)bq * 2 + 0];
    const float ry = qref[(size_t)bq * 2 + 1];
    const float* ob = off + ((size_t)bq * NH + h) * (Ll * Pp * 2);

    float acc = 0.0f;
#pragma unroll
    for (int lvl = 0; lvl < 4; ++lvl) {
        const int Hh = Hs[lvl], Ww = Ws[lvl];
        const float lrx = rx * vrat[(b * Ll + lvl) * 2 + 0];
        const float lry = ry * vrat[(b * Ll + lvl) * 2 + 1];
#pragma unroll
        for (int pt = 0; pt < 4; ++pt) {
            const float ox = ob[(lvl * 4 + pt) * 2 + 0];
            const float oy = ob[(lvl * 4 + pt) * 2 + 1];
            const float locx = lrx + ox / (float)Ww;
            const float locy = lry + oy / (float)Hh;
            const float x = locx * (float)Ww - 0.5f;
            const float y = locy * (float)Hh - 0.5f;
            const float x0 = floorf(x), y0 = floorf(y);
            const float fx = x - x0, fy = y - y0;
            const int x0i = (int)x0, y0i = (int)y0;
            const float aw = p[lvl * 4 + pt] * inv;
            const float wc[4] = {(1.f - fx) * (1.f - fy), fx * (1.f - fy),
                                 (1.f - fx) * fy,         fx * fy};
#pragma unroll
            for (int c = 0; c < 4; ++c) {
                const int xi = x0i + (c & 1);
                const int yi = y0i + (c >> 1);
                if (xi >= 0 && xi < Ww && yi >= 0 && yi < Hh) {   // uniform branch
                    const size_t s = (size_t)St[lvl] + (size_t)yi * Ww + xi;
                    const float vv =
                        value[(((size_t)b * Ss + s) * NH + h) * DH + lane];
                    acc += aw * wc[c] * vv;
                }
            }
        }
    }
    outbf[(size_t)bq * Cc + h * DH + lane] = __float2bfloat16(acc);
}

// ---------------------------------------------------------------------------
// Residual + LayerNorm over C=256; one row per block; optional fp32/bf16 outs.
// ---------------------------------------------------------------------------
__global__ __launch_bounds__(256)
void ln_kernel(const float* __restrict__ r1, const float* __restrict__ r2,
               const float* __restrict__ g, const float* __restrict__ be,
               float* __restrict__ xout, __hip_bfloat16* __restrict__ xbf) {
    __shared__ float s1[256], s2[256];
    const size_t r = blockIdx.x;
    const int t = threadIdx.x;
    const float v = r1[r * Cc + t] + r2[r * Cc + t];
    s1[t] = v; s2[t] = v * v;
    __syncthreads();
    for (int o = 128; o > 0; o >>= 1) {
        if (t < o) { s1[t] += s1[t + o]; s2[t] += s2[t + o]; }
        __syncthreads();
    }
    const float mean = s1[0] * (1.0f / Cc);
    const float var  = s2[0] * (1.0f / Cc) - mean * mean;
    const float o    = (v - mean) * rsqrtf(var + 1e-5f) * g[t] + be[t];
    if (xout) xout[r * Cc + t] = o;
    if (xbf)  xbf[r * Cc + t]  = __float2bfloat16(o);
}

// ---------------------------------------------------------------------------
extern "C" void kernel_launch(void* const* d_in, const int* in_sizes, int n_in,
                              void* d_out, int out_size, void* d_ws, size_t ws_size,
                              hipStream_t stream) {
    const float* src    = (const float*)d_in[0];
    const float* q_feat = (const float*)d_in[1];
    const float* q_pos  = (const float*)d_in[2];
    const float* q_ref  = (const float*)d_in[3];
    const float* vrat   = (const float*)d_in[4];
    const float* Wv     = (const float*)d_in[5];
    const float* bv     = (const float*)d_in[6];
    const float* Woff   = (const float*)d_in[7];
    const float* boff   = (const float*)d_in[8];
    const float* Wa     = (const float*)d_in[9];
    const float* ba     = (const float*)d_in[10];
    const float* Wo     = (const float*)d_in[11];
    const float* bo     = (const float*)d_in[12];
    const float* g1     = (const float*)d_in[13];
    const float* be1    = (const float*)d_in[14];
    const float* W1     = (const float*)d_in[15];
    const float* bf1    = (const float*)d_in[16];
    const float* W2     = (const float*)d_in[17];
    const float* bf2    = (const float*)d_in[18];
    const float* g2     = (const float*)d_in[19];
    const float* be2    = (const float*)d_in[20];
    float* out = (float*)d_out;

    // ---- workspace carve-up (256B aligned) ----
    uintptr_t cur = (uintptr_t)d_ws;
    auto alloc = [&](size_t bytes) {
        uintptr_t p = cur;
        cur = (cur + bytes + 255u) & ~(uintptr_t)255u;
        return (void*)p;
    };
    __hip_bfloat16* srcbf  = (__hip_bfloat16*)alloc((size_t)BS * Cc * 2);
    __hip_bfloat16* qbf    = (__hip_bfloat16*)alloc((size_t)BQ * Cc * 2);
    __hip_bfloat16* Wv_p   = (__hip_bfloat16*)alloc((size_t)Cc * Cc * 2);
    __hip_bfloat16* Woff_p = (__hip_bfloat16*)alloc((size_t)Cc * 256 * 2);
    __hip_bfloat16* Wa_p   = (__hip_bfloat16*)alloc((size_t)Cc * 128 * 2);
    __hip_bfloat16* Wo_p   = (__hip_bfloat16*)alloc((size_t)Cc * Cc * 2);
    __hip_bfloat16* W1_p   = (__hip_bfloat16*)alloc((size_t)Cc * DFF * 2);
    __hip_bfloat16* W2_p   = (__hip_bfloat16*)alloc((size_t)DFF * Cc * 2);
    float*          value  = (float*)alloc((size_t)BS * Cc * 4);
    float*          offb   = (float*)alloc((size_t)BQ * 256 * 4);
    float*          attnl  = (float*)alloc((size_t)BQ * 128 * 4);
    __hip_bfloat16* msbf   = (__hip_bfloat16*)alloc((size_t)BQ * Cc * 2);
    float*          abuf   = (float*)alloc((size_t)BQ * Cc * 4);
    float*          xbuf   = (float*)alloc((size_t)BQ * Cc * 4);
    __hip_bfloat16* xbf    = (__hip_bfloat16*)alloc((size_t)BQ * Cc * 2);
    __hip_bfloat16* hhbf   = (__hip_bfloat16*)alloc((size_t)BQ * DFF * 2);
    float*          fbuf   = (float*)alloc((size_t)BQ * Cc * 4);
    (void)ws_size; (void)n_in; (void)in_sizes; (void)out_size;

    const dim3 blk(256);

    // ---- weight packing (fp32 -> bf16 WMMA B-fragment layout) ----
    pack_b_kernel<<<dim3((Cc * Cc  + 255) / 256), blk, 0, stream>>>(Wv,   Wv_p,   Cc,  Cc);
    pack_b_kernel<<<dim3((Cc * 256 + 255) / 256), blk, 0, stream>>>(Woff, Woff_p, Cc,  256);
    pack_b_kernel<<<dim3((Cc * 128 + 255) / 256), blk, 0, stream>>>(Wa,   Wa_p,   Cc,  128);
    pack_b_kernel<<<dim3((Cc * Cc  + 255) / 256), blk, 0, stream>>>(Wo,   Wo_p,   Cc,  Cc);
    pack_b_kernel<<<dim3((Cc * DFF + 255) / 256), blk, 0, stream>>>(W1,   W1_p,   Cc,  DFF);
    pack_b_kernel<<<dim3((DFF * Cc + 255) / 256), blk, 0, stream>>>(W2,   W2_p,   DFF, Cc);

    // ---- activation conversions ----
    addcvt_kernel<<<dim3((BS * Cc + 255) / 256), blk, 0, stream>>>(src, nullptr, srcbf, BS * Cc);
    addcvt_kernel<<<dim3((BQ * Cc + 255) / 256), blk, 0, stream>>>(q_feat, q_pos, qbf, BQ * Cc);

    // ---- GEMMs (bf16 WMMA, f32 acc) ----
    // value = src @ Wv + bv
    gemm_bf16_kernel<false, false><<<dim3(Cc / 64, (BS + 31) / 32), blk, 0, stream>>>(
        srcbf, Wv_p, bv, value, BS, Cc, Cc);
    // off = q @ Woff + boff
    gemm_bf16_kernel<false, false><<<dim3(256 / 64, (BQ + 31) / 32), blk, 0, stream>>>(
        qbf, Woff_p, boff, offb, BQ, Cc, 256);
    // attn logits = q @ Wa + ba
    gemm_bf16_kernel<false, false><<<dim3(128 / 64, (BQ + 31) / 32), blk, 0, stream>>>(
        qbf, Wa_p, ba, attnl, BQ, Cc, 128);

    // ---- deformable sampling (softmax + bilinear gather) ----
    msdeform_kernel<<<dim3(BQ), blk, 0, stream>>>(value, offb, attnl, q_ref, vrat, msbf);

    // a = msdeform @ Wo + bo
    gemm_bf16_kernel<false, false><<<dim3(Cc / 64, (BQ + 31) / 32), blk, 0, stream>>>(
        msbf, Wo_p, bo, abuf, BQ, Cc, Cc);

    // x = LN(q_feat + a)  (fp32 for final residual + bf16 for FFN GEMM)
    ln_kernel<<<dim3(BQ), blk, 0, stream>>>(q_feat, abuf, g1, be1, xbuf, xbf);

    // hh = relu(x @ W1 + bf1)  -> bf16
    gemm_bf16_kernel<true, true><<<dim3(DFF / 64, (BQ + 31) / 32), blk, 0, stream>>>(
        xbf, W1_p, bf1, hhbf, BQ, Cc, DFF);
    // f = hh @ W2 + bf2
    gemm_bf16_kernel<false, false><<<dim3(Cc / 64, (BQ + 31) / 32), blk, 0, stream>>>(
        hhbf, W2_p, bf2, fbuf, BQ, DFF, Cc);

    // out = LN(x + f)
    ln_kernel<<<dim3(BQ), blk, 0, stream>>>(xbuf, fbuf, g2, be2, out, nullptr);
}
